// LayerNormGRU_46772193853481
// MI455X (gfx1250) — matrix operand order repaired and checked by reading, not compiled, not run on hardware
//
#include <hip/hip_runtime.h>
#include <hip/hip_bf16.h>

#define B_   128
#define S_   1024
#define H_   256
#define G3_  768
#define L_   2
#define EPS_ 1e-5f

// per-step ih slab: 16 rows x 768 f32
#define SLAB_BYTES  (16 * G3_ * 4)          // 49152
#define STEP_BYTES  (B_ * G3_ * 4)          // 393216 (global stride per s)

typedef __attribute__((ext_vector_type(8)))  __bf16 v8bf;
typedef __attribute__((ext_vector_type(16))) __bf16 v16bf;
typedef __attribute__((ext_vector_type(8)))  float  v8f;

union V16U { v16bf v; v8bf h[2]; };

static __device__ __forceinline__ v16bf make_frag(v8bf lo, v8bf hi) {
  V16U u; u.h[0] = lo; u.h[1] = hi; return u.v;
}

// A fragment (16x32 bf16): this lane's row base at the K-tile start.
// lane<16: K = {0..15}; lane>=16: K = {16..31}
static __device__ __forceinline__ v16bf load_a_frag(const __bf16* row_ktile, int khalf) {
  const __bf16* p = row_ktile + khalf * 16;
  return make_frag(*(const v8bf*)p, *(const v8bf*)(p + 8));
}

// B fragment (32x16 bf16): lane holds column (l&15); 16 contiguous K values.
static __device__ __forceinline__ v16bf load_b_frag(const __bf16* p) {
  return make_frag(*(const v8bf*)p, *(const v8bf*)(p + 8));
}

static __device__ __forceinline__ float wave_sum32(float v) {
#pragma unroll
  for (int m = 16; m >= 1; m >>= 1) v += __shfl_xor(v, m, 32);
  return v;
}

static __device__ __forceinline__ float sigmoidf_(float x) {
  return 1.0f / (1.0f + __expf(-x));
}

// Async global->LDS copy of 16 bytes (GVS mode: SGPR64 base + VGPR32 byte offset).
static __device__ __forceinline__ void async_copy_b128(unsigned lds_off,
                                                       int g_off,
                                                       const float* sbase) {
  asm volatile("global_load_async_to_lds_b128 %0, %1, %2"
               :: "v"(lds_off), "v"(g_off), "s"(sbase)
               : "memory");
}

static __device__ __forceinline__ void wait_asynccnt0() {
  asm volatile("s_wait_asynccnt 0x0" ::: "memory");
}

// ---------------------------------------------------------------------------
// Kernel 0: fp32 -> bf16 weight conversion (both layers, both matrices)
// ---------------------------------------------------------------------------
__global__ void __launch_bounds__(256)
cvt_w_kernel(const float* __restrict__ wih, const float* __restrict__ whh,
             __bf16* __restrict__ wihb, __bf16* __restrict__ whhb, int n) {
  int i = blockIdx.x * blockDim.x + threadIdx.x;
  if (i < n) {
    wihb[i] = (__bf16)wih[i];
    whhb[i] = (__bf16)whh[i];
  }
}

// ---------------------------------------------------------------------------
// Kernel 1: ih_all[s,b,g] = sum_d x[b,s,d] * W[g,d]    (bf16 WMMA, f32 accum)
// grid = (2, S): blockIdx.x = batch half (64 rows), blockIdx.y = s
// block = 256 threads (8 waves): wave_m = w&1 (32 rows), wave_n = w>>1 (64 cols)
// ---------------------------------------------------------------------------
__global__ void __launch_bounds__(256)
ih_gemm_kernel(const float* __restrict__ X, const __bf16* __restrict__ W,
               float* __restrict__ OUT) {
  __shared__ __bf16 Ash[64][264];   // 64 rows x 256 K, +8 pad (row = 528B, 16B aligned)

  const int s  = blockIdx.y;
  const int mh = blockIdx.x;
  const int t  = threadIdx.x;

  // Stage x[b, s, 0:256] -> bf16 LDS (4 threads per row, 64 K each)
  {
    const int row  = t >> 2;
    const int koff = (t & 3) * 64;
    const int b    = mh * 64 + row;
    const float* src = X + ((size_t)b * S_ + s) * H_ + koff;
#pragma unroll
    for (int j = 0; j < 64; j += 8) {
      float4 f0 = *(const float4*)(src + j);
      float4 f1 = *(const float4*)(src + j + 4);
      v8bf o;
      o[0] = (__bf16)f0.x; o[1] = (__bf16)f0.y; o[2] = (__bf16)f0.z; o[3] = (__bf16)f0.w;
      o[4] = (__bf16)f1.x; o[5] = (__bf16)f1.y; o[6] = (__bf16)f1.z; o[7] = (__bf16)f1.w;
      *(v8bf*)&Ash[row][koff + j] = o;
    }
  }
  __syncthreads();

  const int wave  = t >> 5, lane = t & 31;
  const int wm    = wave & 1;
  const int wn    = wave >> 1;
  const int lrow  = lane & 15;
  const int khalf = lane >> 4;

  for (int nb = 0; nb < 3; ++nb) {
    v8f acc[2][4];
#pragma unroll
    for (int mf = 0; mf < 2; ++mf)
#pragma unroll
      for (int nf = 0; nf < 4; ++nf) {
        v8f z = {0.f, 0.f, 0.f, 0.f, 0.f, 0.f, 0.f, 0.f};
        acc[mf][nf] = z;
      }

#pragma unroll
    for (int kt = 0; kt < 8; ++kt) {
      v16bf af[2];
#pragma unroll
      for (int mf = 0; mf < 2; ++mf)
        af[mf] = load_a_frag(&Ash[wm * 32 + mf * 16 + lrow][kt * 32], khalf);
      v16bf bfr[4];
#pragma unroll
      for (int nf = 0; nf < 4; ++nf) {
        const int col = nb * 256 + wn * 64 + nf * 16 + lrow;
        bfr[nf] = load_b_frag(W + (size_t)col * H_ + kt * 32 + khalf * 16);
      }
#pragma unroll
      for (int mf = 0; mf < 2; ++mf)
#pragma unroll
        for (int nf = 0; nf < 4; ++nf)
          acc[mf][nf] = __builtin_amdgcn_wmma_f32_16x16x32_bf16(
              false, af[mf], false, bfr[nf], (short)0, acc[mf][nf], false, false);
    }

    // C layout: lane<16 -> M=v, N=lane; lane>=16 -> M=v+8, N=lane-16
#pragma unroll
    for (int mf = 0; mf < 2; ++mf)
#pragma unroll
      for (int nf = 0; nf < 4; ++nf) {
        const int col = nb * 256 + wn * 64 + nf * 16 + lrow;
#pragma unroll
        for (int v = 0; v < 8; ++v) {
          const int b = mh * 64 + wm * 32 + mf * 16 + v + khalf * 8;
          OUT[((size_t)s * B_ + b) * G3_ + col] = acc[mf][nf][v];
        }
      }
  }
}

// ---------------------------------------------------------------------------
// Kernel 2: recurrent scan. grid = 8 (16 batch rows each), block = 768 (24 waves)
// Per step:
//   - all 24 waves: hh = h @ Whh^T via WMMA (Whh fragments register-resident)
//   - waves 0..15:  one batch row each; LN + gates from LDS-resident ih slab
//   - waves 16..23: async global->LDS copy of next step's ih slab (ASYNCcnt)
// ---------------------------------------------------------------------------
__global__ void __launch_bounds__(768)
gru_scan_kernel(const float* __restrict__ IH, const __bf16* __restrict__ Whh,
                const float* __restrict__ hx, const float* __restrict__ gamma,
                const float* __restrict__ beta, float* __restrict__ OUT) {
  __shared__ float  hh[16][G3_];        // 48 KB
  __shared__ __bf16 hbf[16][264];       // 8.25 KB, 16B-aligned rows
  __shared__ float  ihslab[2][16][G3_]; // 2 x 48 KB double-buffered ih tiles

  const int bg    = blockIdx.x;
  const int t     = threadIdx.x;
  const int wave  = t >> 5, lane = t & 31;
  const int lrow  = lane & 15;
  const int khalf = lane >> 4;
  const int d0    = lane * 8;

  float h[8];
  float gr[8], brr[8], gz[8], bz[8], gn[8], bn[8];

  // Uniform global base of this block's ih rows (rows bg*16 .. bg*16+15).
  const float* ih_base = IH + (size_t)bg * 16 * G3_;

  if (wave < 16) {
    const int row = wave;
    const int b   = bg * 16 + row;
#pragma unroll
    for (int j = 0; j < 8; ++j) {
      h[j]   = hx[(size_t)b * H_ + d0 + j];
      gr[j]  = gamma[0 * H_ + d0 + j];  brr[j] = beta[0 * H_ + d0 + j];
      gz[j]  = gamma[1 * H_ + d0 + j];  bz[j]  = beta[1 * H_ + d0 + j];
      gn[j]  = gamma[2 * H_ + d0 + j];  bn[j]  = beta[2 * H_ + d0 + j];
      hbf[row][d0 + j] = (__bf16)h[j];
    }
  } else {
    // Preload slab for s = 0.
    const int idx = (wave - 16) * 32 + lane;                 // 0..255
    const unsigned lds0 = (unsigned)(size_t)&ihslab[0][0][0];
#pragma unroll
    for (int j = 0; j < 12; ++j)
      async_copy_b128(lds0 + idx * 16 + j * 4096, idx * 16 + j * 4096, ih_base);
    wait_asynccnt0();
  }
  __syncthreads();

  const int nt0 = wave * 2;   // 24 waves x 2 tiles = 48 N-tiles = 768 cols

  // Whh B fragments are invariant across all 1024 steps: keep them in VGPRs.
  v16bf bw[16];
#pragma unroll
  for (int kt = 0; kt < 8; ++kt) {
    bw[kt]     = load_b_frag(Whh + (size_t)(nt0 * 16 + lrow) * H_ + kt * 32 + khalf * 16);
    bw[8 + kt] = load_b_frag(Whh + (size_t)((nt0 + 1) * 16 + lrow) * H_ + kt * 32 + khalf * 16);
  }

  for (int s = 0; s < S_; ++s) {
    // ---- hh = h @ Whh^T  (all 24 waves; pure LDS + WMMA) ----
    v8f acc0 = {0.f, 0.f, 0.f, 0.f, 0.f, 0.f, 0.f, 0.f};
    v8f acc1 = {0.f, 0.f, 0.f, 0.f, 0.f, 0.f, 0.f, 0.f};
#pragma unroll
    for (int kt = 0; kt < 8; ++kt) {
      v16bf af = load_a_frag(&hbf[lrow][kt * 32], khalf);
      acc0 = __builtin_amdgcn_wmma_f32_16x16x32_bf16(false, af, false, bw[kt],     (short)0, acc0, false, false);
      acc1 = __builtin_amdgcn_wmma_f32_16x16x32_bf16(false, af, false, bw[8 + kt], (short)0, acc1, false, false);
    }
#pragma unroll
    for (int v = 0; v < 8; ++v) {
      const int m = v + khalf * 8;
      hh[m][nt0 * 16 + lrow]       = acc0[v];
      hh[m][(nt0 + 1) * 16 + lrow] = acc1[v];
    }
    __syncthreads();

    if (wave < 16) {
      // ---- gates: one batch row per wave, ih from LDS slab ----
      const int row = wave;
      const int b   = bg * 16 + row;
      const float* ihrow = &ihslab[s & 1][row][0];

      float r[8], z[8], x[8];
      float s1, s2, mu, inv;

      // r gate
      s1 = 0.f; s2 = 0.f;
#pragma unroll
      for (int j = 0; j < 8; ++j) {
        x[j] = ihrow[d0 + j] + hh[row][d0 + j];
        s1 += x[j]; s2 += x[j] * x[j];
      }
      s1 = wave_sum32(s1); s2 = wave_sum32(s2);
      mu = s1 * (1.0f / H_);
      inv = rsqrtf(s2 * (1.0f / H_) - mu * mu + EPS_);
#pragma unroll
      for (int j = 0; j < 8; ++j)
        r[j] = sigmoidf_((x[j] - mu) * inv * gr[j] + brr[j]);

      // z gate
      s1 = 0.f; s2 = 0.f;
#pragma unroll
      for (int j = 0; j < 8; ++j) {
        x[j] = ihrow[H_ + d0 + j] + hh[row][H_ + d0 + j];
        s1 += x[j]; s2 += x[j] * x[j];
      }
      s1 = wave_sum32(s1); s2 = wave_sum32(s2);
      mu = s1 * (1.0f / H_);
      inv = rsqrtf(s2 * (1.0f / H_) - mu * mu + EPS_);
#pragma unroll
      for (int j = 0; j < 8; ++j)
        z[j] = sigmoidf_((x[j] - mu) * inv * gz[j] + bz[j]);

      // n gate: i_n + r * h_n
      s1 = 0.f; s2 = 0.f;
#pragma unroll
      for (int j = 0; j < 8; ++j) {
        x[j] = ihrow[2 * H_ + d0 + j] + r[j] * hh[row][2 * H_ + d0 + j];
        s1 += x[j]; s2 += x[j] * x[j];
      }
      s1 = wave_sum32(s1); s2 = wave_sum32(s2);
      mu = s1 * (1.0f / H_);
      inv = rsqrtf(s2 * (1.0f / H_) - mu * mu + EPS_);
#pragma unroll
      for (int j = 0; j < 8; ++j) {
        float n = tanhf((x[j] - mu) * inv * gn[j] + bn[j]);
        h[j] = (1.0f - z[j]) * n + z[j] * h[j];
        OUT[((size_t)b * S_ + s) * H_ + d0 + j] = h[j];
        hbf[row][d0 + j] = (__bf16)h[j];
      }
    } else {
      // ---- waves 16..23: async-stage next step's slab into the other buffer ----
      if (s + 1 < S_) {
        const int idx = (wave - 16) * 32 + lane;             // 0..255
        const unsigned ldsb =
            (unsigned)(size_t)&ihslab[(s + 1) & 1][0][0];
        const int gofs = (s + 1) * STEP_BYTES;
#pragma unroll
        for (int j = 0; j < 12; ++j)
          async_copy_b128(ldsb + idx * 16 + j * 4096,
                          gofs + idx * 16 + j * 4096, ih_base);
        wait_asynccnt0();   // retire before the barrier publishes the slab
      }
    }
    __syncthreads();
  }
}

// ---------------------------------------------------------------------------
// Launch
// ---------------------------------------------------------------------------
extern "C" void kernel_launch(void* const* d_in, const int* in_sizes, int n_in,
                              void* d_out, int out_size, void* d_ws, size_t ws_size,
                              hipStream_t stream) {
  const float* x    = (const float*)d_in[0];  // [B,S,H]
  const float* hx   = (const float*)d_in[1];  // [L,B,H]
  const float* Wih  = (const float*)d_in[2];  // [L,3H,H]
  const float* Whh  = (const float*)d_in[3];  // [L,3H,H]
  const float* gam  = (const float*)d_in[4];  // [L,3,H]
  const float* bet  = (const float*)d_in[5];  // [L,3,H]
  float* out = (float*)d_out;                 // [B,S,H]

  char* w = (char*)d_ws;
  float* ihbuf = (float*)w;                               // [S,B,3H] f32 = 384 MiB
  const size_t ihBytes = (size_t)S_ * B_ * G3_ * sizeof(float);
  __bf16* wihb = (__bf16*)(w + ihBytes);                  // [L,3H,H] bf16
  __bf16* whhb = wihb + (size_t)L_ * G3_ * H_;            // [L,3H,H] bf16

  const int nW = L_ * G3_ * H_;
  cvt_w_kernel<<<(nW + 255) / 256, 256, 0, stream>>>(Wih, Whh, wihb, whhb, nW);

  for (int l = 0; l < L_; ++l) {
    const float* xin = (l == 0) ? x : out;
    ih_gemm_kernel<<<dim3(2, S_), 256, 0, stream>>>(
        xin, wihb + (size_t)l * G3_ * H_, ihbuf);
    gru_scan_kernel<<<dim3(B_ / 16), 768, 0, stream>>>(
        ihbuf, whhb + (size_t)l * G3_ * H_,
        hx + (size_t)l * B_ * H_,
        gam + (size_t)l * 3 * H_,
        bet + (size_t)l * 3 * H_, out);
  }
}
